// BertSelfAttention_33586644255284
// MI455X (gfx1250) — matrix operand order repaired and checked by reading, not compiled, not run on hardware
//
#include <hip/hip_runtime.h>
#include <hip/hip_bf16.h>

// BERT self-attention for MI455X (gfx1250): bf16 WMMA, b128 LDS feeds,
// transposed flash attention with register-resident P^T, and TDM
// (tensor_load_to_lds) staging of K blocks. B=4, S=2048, D=768, H=12, DH=64.

typedef __attribute__((ext_vector_type(16))) __bf16 v16bf;
typedef __attribute__((ext_vector_type(8)))  __bf16 v8bf;
typedef __attribute__((ext_vector_type(4)))  __bf16 v4bf;
typedef __attribute__((ext_vector_type(8)))  float  v8f;

typedef unsigned int tdm_v4u __attribute__((ext_vector_type(4)));
typedef int          tdm_v8i __attribute__((ext_vector_type(8)));
typedef int          tdm_v4i __attribute__((ext_vector_type(4)));

#define Bn 4
#define Sn 2048
#define Dn 768
#define Hn 12
#define DHn 64

#if defined(__has_builtin)
#if __has_builtin(__builtin_amdgcn_tensor_load_to_lds)
#define HAVE_TDM 1
#endif
#endif

union V16U { v16bf v; v8bf h[2]; unsigned u[8]; };

// fp32 -> bf16 via hardware convert (v_cvt_pk_bf16_f32)
static __device__ inline __bf16 f2bf(float f) { return static_cast<__bf16>(f); }
static __device__ inline unsigned short f2bfu(float f) {
    return __builtin_bit_cast(unsigned short, static_cast<__bf16>(f));
}
static __device__ inline v8f v8f_zero() {
    v8f z = {0.f, 0.f, 0.f, 0.f, 0.f, 0.f, 0.f, 0.f};
    return z;
}

#ifdef HAVE_TDM
// TDM: DMA a 64x64 bf16 tile (row stride 64 elems in memory) into LDS with
// 16B padding after each 128B row -> matches __bf16 [64][72] layout.
static __device__ inline void tdm_load_k_tile(unsigned ldsAddr,
                                              unsigned long long gaddr) {
    tdm_v4u g0;
    g0[0] = 1u;                                       // count=1, user mode
    g0[1] = ldsAddr;                                  // lds_addr (bytes)
    g0[2] = (unsigned)(gaddr & 0xFFFFFFFFull);        // global_addr[31:0]
    g0[3] = (unsigned)((gaddr >> 32) & 0x01FFFFFFull) // global_addr[56:32]
          | (2u << 30);                               // type = 2 ("image")
    tdm_v8i g1;
    g1[0] = (1 << 16)    // data_size = 1 -> 2 bytes
          | (1 << 20)    // pad_enable
          | (4 << 22)    // pad_interval code 4 -> every 32 DWORDs (128B)
          | (3 << 25);   // pad_amount  code 3 -> 4 DWORDs (16B)
    g1[1] = (int)((DHn & 0xFFFF) << 16);   // tensor_dim0 = 64 (low 16)
    g1[2] = (int)((Sn & 0xFFFF) << 16);    // dim0 hi=0 | tensor_dim1 = 2048 lo
    g1[3] = (int)(64 << 16);               // dim1 hi=0 | tile_dim0 = 64
    g1[4] = 64;                            // tile_dim1 = 64 (tile_dim2 = 0)
    g1[5] = DHn;                           // tensor_dim0_stride = 64 (lo 32)
    g1[6] = 0;                             // dim0_stride hi | dim1_stride lo
    g1[7] = 0;
    tdm_v4i z4 = {0, 0, 0, 0};
#if __clang_major__ >= 23
    tdm_v8i z8 = {0, 0, 0, 0, 0, 0, 0, 0};
    __builtin_amdgcn_tensor_load_to_lds(g0, g1, z4, z4, z8, 0);
#else
    __builtin_amdgcn_tensor_load_to_lds(g0, g1, z4, z4, 0);
#endif
}
#endif

// ---------------------------------------------------------------------------
// Kernel 1: fused QKV projection. out[z] = x @ W[z] + b[z] -> bf16 [B,H,S,DH].
// Q pre-scaled by 1/sqrt(DH). Grid (M/128, N/64, 3), block 256 (8 waves).
// ---------------------------------------------------------------------------
__global__ __launch_bounds__(256) void qkv_proj_kernel(
    const float* __restrict__ x,
    const float* __restrict__ Wq, const float* __restrict__ bq,
    const float* __restrict__ Wk, const float* __restrict__ bk,
    const float* __restrict__ Wv, const float* __restrict__ bv,
    unsigned short* __restrict__ Qb,
    unsigned short* __restrict__ Kb,
    unsigned short* __restrict__ Vb)
{
    __shared__ __bf16 As[128][40];   // 128 rows x 32 K (+pad, 80B row stride)
    __shared__ __bf16 Bs[64][40];    // 64 cols x 32 K, K-major (transposed W)

    const int tid  = threadIdx.x;
    const int wave = tid >> 5;
    const int lane = tid & 31;
    const int nn   = lane & 15;
    const int hi16 = lane >> 4;
    const int kbA  = hi16 ? 8 : 0;
    const int kbB  = hi16 ? 16 : 0;

    const int mBase = blockIdx.x * 128;
    const int nBase = blockIdx.y * 64;
    const int z     = blockIdx.z;

    const float* W    = (z == 0) ? Wq : (z == 1) ? Wk : Wv;
    const float* bias = (z == 0) ? bq : (z == 1) ? bk : bv;
    unsigned short* dst = (z == 0) ? Qb : (z == 1) ? Kb : Vb;
    const float oscale = (z == 0) ? 0.125f : 1.0f;

    const int wn  = tid & 63;
    const int wk0 = (tid >> 6) * 8;

    v8f acc[4];
    #pragma unroll
    for (int t = 0; t < 4; ++t) acc[t] = v8f_zero();

    for (int kc = 0; kc < Dn; kc += 32) {
        #pragma unroll
        for (int i = 0; i < 4; ++i) {
            int idx = tid + i * 256;
            int r   = idx >> 3;
            int c4  = (idx & 7) * 4;
            float4 f = *(const float4*)&x[(size_t)(mBase + r) * Dn + kc + c4];
            v4bf p;
            p[0] = f2bf(f.x); p[1] = f2bf(f.y); p[2] = f2bf(f.z); p[3] = f2bf(f.w);
            *(v4bf*)&As[r][c4] = p;
        }
        {
            v8bf p;
            #pragma unroll
            for (int j = 0; j < 8; ++j)
                p[j] = f2bf(W[(size_t)(kc + wk0 + j) * Dn + nBase + wn]);
            *(v8bf*)&Bs[wn][wk0] = p;
        }
        __syncthreads();

        V16U a;
        a.h[0] = *(const v8bf*)&As[16 * wave + nn][kbA];
        a.h[1] = *(const v8bf*)&As[16 * wave + nn][kbA + 16];
        #pragma unroll
        for (int t = 0; t < 4; ++t) {
            V16U bfr;
            bfr.h[0] = *(const v8bf*)&Bs[t * 16 + nn][kbA];
            bfr.h[1] = *(const v8bf*)&Bs[t * 16 + nn][kbA + 16];
            acc[t] = __builtin_amdgcn_wmma_f32_16x16x32_bf16(
                false, a.v, false, bfr.v, (short)0, acc[t], false, false);
        }
        __syncthreads();
    }

    const int mRow0 = mBase + 16 * wave + (hi16 ? 8 : 0);
    #pragma unroll
    for (int t = 0; t < 4; ++t) {
        int col = nBase + t * 16 + nn;
        int h  = col >> 6;
        int dh = col & 63;
        float bb = bias[col];
        #pragma unroll
        for (int r = 0; r < 8; ++r) {
            int row = mRow0 + r;
            int bidx = row >> 11;
            int sidx = row & 2047;
            float v = (acc[t][r] + bb) * oscale;
            dst[(((size_t)(bidx * Hn + h) * Sn + sidx) * DHn) + dh] = f2bfu(v);
        }
    }
}

// ---------------------------------------------------------------------------
// Kernel 2: transposed flash attention. Grid (S/128, B*H), block 256 (8 waves).
//   S^T(tile) = K(16key x 32dh) x Q^T(32dh x 16q)   [A=K from Kt, B=Q global]
//   O^T(tile) = V^T(16dh x 32key) x P^T(32key x 16q)[A=V^T from Vt, B=P regs]
// K block staged by the Tensor Data Mover (with pad -> [64][72] layout).
// ---------------------------------------------------------------------------
__global__ __launch_bounds__(256) void attn_kernel(
    const unsigned short* __restrict__ Qb,
    const unsigned short* __restrict__ Kb,
    const unsigned short* __restrict__ Vb,
    const float* __restrict__ mask,       // [B, S] additive
    float* __restrict__ out)              // [B, S, D]
{
    __shared__ __bf16 Kt[64][72];         // 64 keys x 64 dh (+16B row pad)
    __shared__ __bf16 Vt[64][72];         // 64 dh x 64 keys (transposed, +pad)

    const int tid  = threadIdx.x;
    const int wave = tid >> 5;
    const int lane = tid & 31;
    const int nn   = lane & 15;
    const int hi16 = lane >> 4;
    const int kbA  = hi16 ? 8 : 0;       // A-frag K base
    const int kbB  = hi16 ? 16 : 0;      // B-frag K base

    const int bh = blockIdx.y;
    const int b  = bh / Hn;
    const int h  = bh % Hn;
    const int qBase = blockIdx.x * 128;
    const int q     = qBase + wave * 16 + nn;   // this lane's query row

    const size_t headOff = (size_t)bh * Sn * DHn;

    const int vdh = tid & 63;
    const int vk0 = (tid >> 6) * 16;

#ifdef HAVE_TDM
    const unsigned ldsK = (unsigned)(size_t)&Kt[0][0];
    const unsigned long long kBase =
        (unsigned long long)(size_t)Kb + 2ull * headOff;
#endif

    // Q as B-fragments: two 32-dh chunks, each 16 contiguous bf16 per lane
    v16bf bq2[2];
    {
        const size_t qrow = headOff + (size_t)q * DHn;
        #pragma unroll
        for (int kc = 0; kc < 2; ++kc) {
            V16U u;
            u.h[0] = *(const v8bf*)&Qb[qrow + kc * 32 + kbB];
            u.h[1] = *(const v8bf*)&Qb[qrow + kc * 32 + kbB + 8];
            bq2[kc] = u.v;
        }
    }

    v8f o[4];
    #pragma unroll
    for (int t = 0; t < 4; ++t) o[t] = v8f_zero();
    float mrun = -3.0e38f, lrun = 0.f;

    for (int j = 0; j < Sn; j += 64) {
        __syncthreads();   // previous iteration's readers done with Kt/Vt
#ifdef HAVE_TDM
        // K block via Tensor Data Mover (wave 0 issues; EXEC ignored by TDM)
        if (wave == 0)
            tdm_load_k_tile(ldsK, kBase + 2ull * (unsigned long long)j * DHn);
#else
        // K block: straight 16B copies, row-major [key][dh]
        #pragma unroll
        for (int i = 0; i < 2; ++i) {
            int idx = tid + i * 256;
            int r   = idx >> 3;
            int c8  = (idx & 7) * 8;
            *(v8bf*)&Kt[r][c8] =
                *(const v8bf*)&Kb[headOff + (size_t)(j + r) * DHn + c8];
        }
#endif
        // V block transposed: Vt[dh][key]; coalesced reads per key row
        #pragma unroll
        for (int g = 0; g < 2; ++g) {
            v8bf p;
            #pragma unroll
            for (int jj = 0; jj < 8; ++jj)
                p[jj] = __builtin_bit_cast(__bf16,
                    Vb[headOff + (size_t)(j + vk0 + g * 8 + jj) * DHn + vdh]);
            *(v8bf*)&Vt[vdh][vk0 + g * 8] = p;
        }
#ifdef HAVE_TDM
        if (wave == 0) __builtin_amdgcn_s_wait_tensorcnt(0);
#endif
        __syncthreads();

        // ---- S^T: lane = query column; rows = keys (4 tiles of 16)
        v8f s[4];
        #pragma unroll
        for (int t = 0; t < 4; ++t) {
            s[t] = v8f_zero();
            #pragma unroll
            for (int kc = 0; kc < 2; ++kc) {
                V16U af;   // A = K tile rows: contiguous dh groups in Kt
                af.h[0] = *(const v8bf*)&Kt[t * 16 + nn][kc * 32 + kbB];
                af.h[1] = *(const v8bf*)&Kt[t * 16 + nn][kc * 32 + kbB + 8];
                s[t] = __builtin_amdgcn_wmma_f32_16x16x32_bf16(
                    false, af.v, false, bq2[kc], (short)0, s[t], false, false);
            }
            // additive mask varies along keys: lane's keys t*16+hi16*8+r
            const float4* mp =
                (const float4*)&mask[(size_t)b * Sn + j + t * 16 + hi16 * 8];
            float4 m0 = mp[0], m1 = mp[1];
            s[t][0] += m0.x; s[t][1] += m0.y; s[t][2] += m0.z; s[t][3] += m0.w;
            s[t][4] += m1.x; s[t][5] += m1.y; s[t][6] += m1.z; s[t][7] += m1.w;
        }

        // ---- online softmax: per-lane (query) over 32 in-lane keys + merge
        float bm = s[0][0];
        #pragma unroll
        for (int t = 0; t < 4; ++t)
            #pragma unroll
            for (int r = 0; r < 8; ++r) bm = fmaxf(bm, s[t][r]);
        bm = fmaxf(bm, __shfl_xor(bm, 16, 32));
        float mnew = fmaxf(mrun, bm);
        float sc   = __expf(mrun - mnew);
        float rs   = 0.f;
        #pragma unroll
        for (int t = 0; t < 4; ++t)
            #pragma unroll
            for (int r = 0; r < 8; ++r) {
                float p = __expf(s[t][r] - mnew);
                s[t][r] = p;
                rs += p;
            }
        rs += __shfl_xor(rs, 16, 32);
        lrun = lrun * sc + rs;
        mrun = mnew;
        #pragma unroll
        for (int t = 0; t < 4; ++t) o[t] = o[t] * sc;

        // ---- P^T in registers: pack to bf16 pairs, half-swap via shfl_xor(16)
        unsigned pk[4][4], px[4][4];
        #pragma unroll
        for (int t = 0; t < 4; ++t)
            #pragma unroll
            for (int i = 0; i < 4; ++i) {
                pk[t][i] = (unsigned)f2bfu(s[t][2 * i]) |
                           ((unsigned)f2bfu(s[t][2 * i + 1]) << 16);
                px[t][i] = __shfl_xor(pk[t][i], 16, 32);
            }

        // ---- O^T += V^T x P^T over two 32-key chunks
        #pragma unroll
        for (int c = 0; c < 2; ++c) {
            const int t0 = 2 * c, t1 = 2 * c + 1;
            V16U pb;   // B-frag of P^T: low half keys c*32+0..15, high +16..31
            #pragma unroll
            for (int i = 0; i < 4; ++i) {
                pb.u[i]     = hi16 ? px[t1][i] : pk[t0][i];
                pb.u[4 + i] = hi16 ? pk[t1][i] : px[t0][i];
            }
            #pragma unroll
            for (int t = 0; t < 4; ++t) {
                V16U vf;   // A = V^T tile rows: contiguous keys in Vt
                vf.h[0] = *(const v8bf*)&Vt[t * 16 + nn][c * 32 + kbB];
                vf.h[1] = *(const v8bf*)&Vt[t * 16 + nn][c * 32 + kbB + 8];
                o[t] = __builtin_amdgcn_wmma_f32_16x16x32_bf16(
                    false, vf.v, false, pb.v, (short)0, o[t], false, false);
            }
        }
    }

    // ---- write ctx: lane holds 8 consecutive dh per tile -> float4 stores
    const float inv = 1.0f / lrun;
    const size_t rowBase = (size_t)(b * Sn + q) * Dn + h * DHn + hi16 * 8;
    #pragma unroll
    for (int t = 0; t < 4; ++t) {
        float4 w0, w1;
        w0.x = o[t][0] * inv; w0.y = o[t][1] * inv;
        w0.z = o[t][2] * inv; w0.w = o[t][3] * inv;
        w1.x = o[t][4] * inv; w1.y = o[t][5] * inv;
        w1.z = o[t][6] * inv; w1.w = o[t][7] * inv;
        *(float4*)&out[rowBase + t * 16]     = w0;
        *(float4*)&out[rowBase + t * 16 + 4] = w1;
    }
}

extern "C" void kernel_launch(void* const* d_in, const int* in_sizes, int n_in,
                              void* d_out, int out_size, void* d_ws, size_t ws_size,
                              hipStream_t stream) {
    (void)in_sizes; (void)n_in; (void)out_size; (void)ws_size;
    const float* hidden = (const float*)d_in[0];
    const float* mask   = (const float*)d_in[1];
    const float* Wq     = (const float*)d_in[2];
    const float* bq     = (const float*)d_in[3];
    const float* Wk     = (const float*)d_in[4];
    const float* bk     = (const float*)d_in[5];
    const float* Wv     = (const float*)d_in[6];
    const float* bv     = (const float*)d_in[7];
    float* out = (float*)d_out;

    const size_t nQKV = (size_t)Bn * Hn * Sn * DHn;
    unsigned short* Qb = (unsigned short*)d_ws;
    unsigned short* Kb = Qb + nQKV;
    unsigned short* Vb = Kb + nQKV;

    qkv_proj_kernel<<<dim3(Bn * Sn / 128, Dn / 64, 3), dim3(256), 0, stream>>>(
        hidden, Wq, bq, Wk, bk, Wv, bv, Qb, Kb, Vb);
    attn_kernel<<<dim3(Sn / 128, Bn * Hn), dim3(256), 0, stream>>>(
        Qb, Kb, Vb, mask, out);
}
